// DeepHamCritic_35012573397744
// MI455X (gfx1250) — matrix operand order repaired and checked by reading, not compiled, not run on hardware
//
#include <hip/hip_runtime.h>

// ---------------------------------------------------------------------------
// GATv2 (3 layers) + MLP head for MI455X (gfx1250, wave32, WMMA, TDM).
// - GEMMs: v_wmma_f32_16x16x32_bf16, f32 accumulate.
//   * B (weights) pre-packed into WMMA fragment-native layout -> direct
//     global_load_b128 fragment fetches, no LDS transpose.
//   * A (activations) staged to LDS by the Tensor Data Mover
//     (tensor_load_to_lds, double-buffered, s_wait_tensorcnt).
// - Edge softmax/aggregation: wave-per-edge coalesced gathers; xs/xt/agg are
//   20MB each -> resident in 192MB L2; f32 global atomics land in L2.
// ---------------------------------------------------------------------------

#define NN   10000      // nodes
#define NE   160000     // edges (without self loops)
#define NET  170000     // edges + self loops
#define DIN  128
#define DH   512

typedef __attribute__((ext_vector_type(16))) __bf16        bf16x16;
typedef __attribute__((ext_vector_type(8)))  float         floatx8;
typedef __attribute__((ext_vector_type(4)))  unsigned int  uint32x4;
typedef __attribute__((ext_vector_type(8)))  int           int32x8;
typedef __attribute__((ext_vector_type(4)))  int           int32x4;

static __device__ __forceinline__ unsigned short f32_to_bf16_rne(float f) {
    unsigned int u = __float_as_uint(f);
    unsigned int r = u + 0x7fffu + ((u >> 16) & 1u);
    return (unsigned short)(r >> 16);
}

// f32 -> bf16 conversion (GEMM A-side activations)
__global__ __launch_bounds__(256) void cvt_bf16_kernel(
    const float* __restrict__ in, unsigned short* __restrict__ out, int n) {
    int i = blockIdx.x * 256 + threadIdx.x;
    if (i < n) out[i] = f32_to_bf16_rne(in[i]);
}

// f32 W[K][N] -> bf16 packed in WMMA B-fragment-native layout.
// Fragment block (kt=k/32, nt=n/16); within a block, each column n' (0..15)
// stores its 32 K-values contiguously (64B). A wave's B-fragment load is then
// 2x global_load_b128 per lane:
//   addr = (((kt*(N/16) + nt)*16 + (n&15))*32 + (k&31))
__global__ __launch_bounds__(256) void cvt_pack_b_kernel(
    const float* __restrict__ in, unsigned short* __restrict__ out, int K, int N) {
    int i = blockIdx.x * 256 + threadIdx.x;
    if (i >= K * N) return;
    int k = i / N, n = i - k * N;
    size_t o = (((size_t)(k >> 5) * (N >> 4) + (n >> 4)) * 16 + (n & 15)) * 32 + (k & 31);
    out[o] = f32_to_bf16_rne(in[i]);
}

// ---------------------------------------------------------------------------
// Tiled WMMA GEMM:  C[M,N] = A[M,K](bf16) * B(packed bf16) + bias, opt leaky.
// Block: 256 threads = 8 waves. Block tile 128(M) x 64(N); wave tile 32x32
// (2x2 of v_wmma_f32_16x16x32_bf16). K stepped by 32.
// A tile (128x32 bf16, 8KB) staged to LDS by the TDM, double buffered.
// ---------------------------------------------------------------------------
#define TM 128
#define TN 64
#define TK 32

// Issue a TDM 2D tile load: A[m0 .. m0+rows)[k0 .. k0+32) -> LDS @ lds_off.
// D# per CDNA5 ISA §8.3/8.4: group0 = {count=1 | lds_addr | global_addr | type=2},
// group1 = {data_size=2B, tensor_dim0=K, tensor_dim1=M, tile=32 x rows,
//           tensor_dim0_stride=K}; groups 2/3 zero (2D tensor).
static __device__ __forceinline__ void tdm_load_a_tile(
    const unsigned short* A, unsigned int lds_off, int M, int K, int m0, int k0)
{
    unsigned long long ga =
        (unsigned long long)(size_t)(const void*)(A + (size_t)m0 * K + k0);
    int rows = M - m0; if (rows > TM) rows = TM;

    uint32x4 g0;
    g0[0] = 1u;                                            // count=1 (user D#)
    g0[1] = lds_off;                                       // lds_addr (bytes)
    g0[2] = (unsigned int)(ga & 0xffffffffull);            // global_addr[31:0]
    g0[3] = (unsigned int)((ga >> 32) & 0x01ffffffull)     // global_addr[56:32]
          | 0x80000000u;                                   // type=2 ("image")

    int32x8 g1;
    g1[0] = 0x00010000;                                    // data_size=1 -> 2B
    g1[1] = (K & 0xffff) << 16;                            // tensor_dim0[15:0]
    g1[2] = ((unsigned)K >> 16) | ((M & 0xffff) << 16);    // dim0[31:16]|dim1[15:0]
    g1[3] = ((unsigned)M >> 16) | (TK << 16);              // dim1[31:16]|tile_dim0=32
    g1[4] = rows;                                          // tile_dim1 (tile_dim2=0)
    g1[5] = K;                                             // tensor_dim0_stride[31:0]
    g1[6] = 0;                                             // stride0[47:32]|stride1[15:0]
    g1[7] = 0;                                             // stride1[47:16]

    int32x4 gz = {0, 0, 0, 0};
#if __clang_major__ >= 23
    int32x8 gz8 = {0, 0, 0, 0, 0, 0, 0, 0};
    __builtin_amdgcn_tensor_load_to_lds(g0, g1, gz, gz, gz8, 0);
#else
    __builtin_amdgcn_tensor_load_to_lds(g0, g1, gz, gz, 0);
#endif
}

__global__ __launch_bounds__(256) void gemm_bf16_wmma_kernel(
    const unsigned short* __restrict__ A,   // [M][K] bf16 bits, row major
    const unsigned short* __restrict__ Bp,  // packed B fragments (see packer)
    const float* __restrict__ bias,         // [N]
    float* __restrict__ C,                  // [M][N] f32
    int M, int N, int K, int act)           // act: 0=none, 1=leakyReLU(0.01)
{
    __shared__ unsigned short ldsA[2][TM * TK];   // double-buffered A tile

    const int tid  = threadIdx.x;
    const int lane = tid & 31;
    const int wave = tid >> 5;
    const int m0 = blockIdx.x * TM;
    const int n0 = blockIdx.y * TN;
    const int wm = (wave & 3) * 32;        // 4 wave rows
    const int wn = (wave >> 2) * 32;       // 2 wave cols
    const int g  = lane >> 4;
    const int lm = lane & 15;
    const int nK = K / TK;
    const int nTiles16 = N >> 4;

    union Frag { bf16x16 v; uint4 q[2]; };

    floatx8 acc[2][2];
    const floatx8 fzero = {0.f,0.f,0.f,0.f,0.f,0.f,0.f,0.f};
    acc[0][0] = fzero; acc[0][1] = fzero; acc[1][0] = fzero; acc[1][1] = fzero;

    // prime the pipeline: TDM fetch of K-tile 0 into buffer 0
    if (wave == 0)
        tdm_load_a_tile(A, (unsigned int)(size_t)(void*)&ldsA[0][0], M, K, m0, 0);

    for (int kt = 0; kt < nK; ++kt) {
        const int buf = kt & 1;
        // buffer `buf` ready (TENSORcnt==0) + all waves done with prev reads
        if (wave == 0) __builtin_amdgcn_s_wait_tensorcnt(0);
        __syncthreads();
        // prefetch next K-tile into the other buffer while we compute
        if (wave == 0 && kt + 1 < nK)
            tdm_load_a_tile(A, (unsigned int)(size_t)(void*)&ldsA[buf ^ 1][0],
                            M, K, m0, (kt + 1) * TK);

        // A fragments (ISA 7.12.2, 16x32 bf16): lane -> row lm, K-half g;
        // v0..3 = K(8g..8g+7), v4..7 = K(16+8g..) -> 2x ds_load_b128
        Frag a[2], b[2];
        #pragma unroll
        for (int i = 0; i < 2; ++i) {
            const unsigned short* base = &ldsA[buf][(wm + i * 16 + lm) * TK + g * 8];
            a[i].q[0] = *(const uint4*)(base);
            a[i].q[1] = *(const uint4*)(base + 16);
        }
        // B fragments straight from packed global (L2-resident weights):
        // lane -> col lm, K-half g; v0..3 = K(8g..8g+7), v4..7 = K(16+8g..) -> 2x b128
        #pragma unroll
        for (int j = 0; j < 2; ++j) {
            const unsigned short* base =
                Bp + (((size_t)kt * nTiles16 + ((n0 + wn) >> 4) + j) * 16 + lm) * 32 + g * 8;
            b[j].q[0] = *(const uint4*)(base);
            b[j].q[1] = *(const uint4*)(base + 16);
        }
        #pragma unroll
        for (int i = 0; i < 2; ++i)
            #pragma unroll
            for (int j = 0; j < 2; ++j)
                acc[i][j] = __builtin_amdgcn_wmma_f32_16x16x32_bf16(
                    false, a[i].v, false, b[j].v, (short)0, acc[i][j], false, false);
    }

    // epilogue: bias + optional leakyReLU(0.01); D layout per ISA 7.12.2
    #pragma unroll
    for (int j = 0; j < 2; ++j) {
        int n = n0 + wn + j * 16 + lm;
        float bv = bias[n];
        #pragma unroll
        for (int i = 0; i < 2; ++i) {
            #pragma unroll
            for (int r = 0; r < 8; ++r) {
                int m = m0 + wm + i * 16 + r + 8 * g;
                if (m < M) {
                    float v = acc[i][j][r] + bv;
                    if (act == 1) v = v > 0.f ? v : 0.01f * v;
                    C[(size_t)m * N + n] = v;
                }
            }
        }
    }
}

// init per-layer segment state: mmax = mapped(-inf), ssum = 0, agg = 0
__global__ __launch_bounds__(256) void seg_init_kernel(
    int* __restrict__ mmax, float* __restrict__ ssum, float* __restrict__ agg) {
    int i = blockIdx.x * 256 + threadIdx.x;
    if (i < NN) { mmax[i] = (int)0x80000000; ssum[i] = 0.f; }
    if (i < NN * DH) agg[i] = 0.f;
}

// pass 1: e = leaky_relu(xs[src]+xt[dst], 0.2) . a ; segment max via int-mapped
// atomicMax. One wave per edge; 2KB rows gathered with float4 lane strides.
__global__ __launch_bounds__(256) void edge_score_kernel(
    const int* __restrict__ ei, const float* __restrict__ xs,
    const float* __restrict__ xt, const float* __restrict__ avec,
    float* __restrict__ esc, int* __restrict__ mmax)
{
    int wave = threadIdx.x >> 5;
    int lane = threadIdx.x & 31;
    int e = blockIdx.x * 8 + wave;
    if (e >= NET) return;
    int s, d;
    if (e < NE) { s = ei[e]; d = ei[NE + e]; } else { s = d = e - NE; }
    const float* xr = xs + (size_t)s * DH;
    const float* tr = xt + (size_t)d * DH;
    __builtin_prefetch(xr + lane * 16, 0, 0);   // global_prefetch_b8: warm rows
    __builtin_prefetch(tr + lane * 16, 0, 0);
    float sum = 0.f;
    #pragma unroll
    for (int c = 0; c < 4; ++c) {
        int base = c * 128 + lane * 4;
        float4 xv = *(const float4*)(xr + base);
        float4 tv = *(const float4*)(tr + base);
        float4 av = *(const float4*)(avec + base);
        float v0 = xv.x + tv.x; v0 = v0 > 0.f ? v0 : 0.2f * v0;
        float v1 = xv.y + tv.y; v1 = v1 > 0.f ? v1 : 0.2f * v1;
        float v2 = xv.z + tv.z; v2 = v2 > 0.f ? v2 : 0.2f * v2;
        float v3 = xv.w + tv.w; v3 = v3 > 0.f ? v3 : 0.2f * v3;
        sum += v0 * av.x + v1 * av.y + v2 * av.z + v3 * av.w;
    }
    #pragma unroll
    for (int o = 16; o > 0; o >>= 1) sum += __shfl_xor(sum, o, 32);
    if (lane == 0) {
        esc[e] = sum;
        int iv = __float_as_int(sum);
        iv = iv >= 0 ? iv : (iv ^ 0x7fffffff);      // order-preserving map
        atomicMax(&mmax[d], iv);
    }
}

// pass 2: ex = exp(e - m[dst]); s[dst] += ex
__global__ __launch_bounds__(256) void edge_exp_kernel(
    const int* __restrict__ ei, const float* __restrict__ esc,
    const int* __restrict__ mmax, float* __restrict__ exb,
    float* __restrict__ ssum)
{
    int e = blockIdx.x * 256 + threadIdx.x;
    if (e >= NET) return;
    int d = (e < NE) ? ei[NE + e] : (e - NE);
    int mv = mmax[d];
    float m = __int_as_float(mv >= 0 ? mv : (mv ^ 0x7fffffff));
    float v = __expf(esc[e] - m);
    exb[e] = v;
    atomicAdd(&ssum[d], v);
}

// pass 3: agg[dst] += (ex/s[dst]) * xs[src]; wave per edge, 512 f32 atomics
// (agg is 20MB -> L2-resident; global_atomic_add_f32, no return)
__global__ __launch_bounds__(256) void edge_aggregate_kernel(
    const int* __restrict__ ei, const float* __restrict__ exb,
    const float* __restrict__ ssum, const float* __restrict__ xs,
    float* __restrict__ agg)
{
    int wave = threadIdx.x >> 5;
    int lane = threadIdx.x & 31;
    int e = blockIdx.x * 8 + wave;
    if (e >= NET) return;
    int s, d;
    if (e < NE) { s = ei[e]; d = ei[NE + e]; } else { s = d = e - NE; }
    const float* xr = xs + (size_t)s * DH;
    __builtin_prefetch(xr + lane * 16, 0, 0);
    float alpha = exb[e] / ssum[d];
    float* ar = agg + (size_t)d * DH;
    #pragma unroll
    for (int c = 0; c < 16; ++c) {
        int i = c * 32 + lane;
        atomicAdd(&ar[i], alpha * xr[i]);
    }
}

// h = tanh(agg + c), written directly as bf16 for the next layer's GEMM A
__global__ __launch_bounds__(256) void tanh_cvt_kernel(
    const float* __restrict__ agg, const float* __restrict__ cvec,
    unsigned short* __restrict__ hbf)
{
    int i = blockIdx.x * 256 + threadIdx.x;
    if (i >= NN * DH) return;
    float v = tanhf(agg[i] + cvec[i & (DH - 1)]);
    hbf[i] = f32_to_bf16_rne(v);
}

// out[m] = h[m,:] . Wh3 + bh3  (one wave per row)
__global__ __launch_bounds__(256) void final_dot_kernel(
    const float* __restrict__ h, const float* __restrict__ w,
    const float* __restrict__ b, float* __restrict__ out)
{
    int wave = threadIdx.x >> 5;
    int lane = threadIdx.x & 31;
    int m = blockIdx.x * 8 + wave;
    if (m >= NN) return;
    const float* hr = h + (size_t)m * DH;
    float sum = 0.f;
    #pragma unroll
    for (int c = 0; c < 16; ++c) {
        int i = c * 32 + lane;
        sum += hr[i] * w[i];
    }
    #pragma unroll
    for (int o = 16; o > 0; o >>= 1) sum += __shfl_xor(sum, o, 32);
    if (lane == 0) out[m] = sum + b[0];
}

// ---------------------------------------------------------------------------
extern "C" void kernel_launch(void* const* d_in, const int* in_sizes, int n_in,
                              void* d_out, int out_size, void* d_ws, size_t ws_size,
                              hipStream_t stream) {
    (void)in_sizes; (void)n_in; (void)out_size; (void)ws_size;

    const float* x  = (const float*)d_in[0];
    const int*   ei = (const int*)d_in[1];
    // layer params: l in {0,1,2} at base 2+6l: Ws, bs, Wt, bt, a, c
    const float* Wh1 = (const float*)d_in[20]; const float* bh1 = (const float*)d_in[21];
    const float* Wh2 = (const float*)d_in[22]; const float* bh2 = (const float*)d_in[23];
    const float* Wh3 = (const float*)d_in[24]; const float* bh3 = (const float*)d_in[25];
    float* out = (float*)d_out;

    // carve workspace (~74 MB total)
    char* ws = (char*)d_ws;
    size_t off = 0;
    auto carve = [&](size_t bytes) -> void* {
        void* p = ws + off;
        off = (off + bytes + 255) & ~(size_t)255;
        return p;
    };
    unsigned short* hbf  = (unsigned short*)carve((size_t)NN * DH * 2); // GEMM A (bf16)
    unsigned short* wS   = (unsigned short*)carve((size_t)DH * DH * 2); // packed B (src/shared)
    unsigned short* wT   = (unsigned short*)carve((size_t)DH * DH * 2); // packed B (tgt)
    float* xs   = (float*)carve((size_t)NN * DH * 4);
    float* xt   = (float*)carve((size_t)NN * DH * 4);
    float* agg  = (float*)carve((size_t)NN * DH * 4);
    float* esc  = (float*)carve((size_t)NET * 4);
    float* exb  = (float*)carve((size_t)NET * 4);
    int*   mmax = (int*)carve((size_t)NN * 4);
    float* ssum = (float*)carve((size_t)NN * 4);

    const dim3 blk(256);
    const dim3 gemmGrid((NN + TM - 1) / TM, DH / TN);
    const int nNDH   = NN * DH;
    const int gNDH   = (nNDH + 255) / 256;
    const int gEdgeW = (NET + 7) / 8;      // wave-per-edge kernels
    const int gEdgeT = (NET + 255) / 256;  // thread-per-edge kernels

    // x -> bf16 (layer 1 GEMM A input)
    cvt_bf16_kernel<<<(NN * DIN + 255) / 256, blk, 0, stream>>>(x, hbf, NN * DIN);

    for (int l = 0; l < 3; ++l) {
        const int base = 2 + 6 * l;
        const float* Wsrc = (const float*)d_in[base + 0];
        const float* bsrc = (const float*)d_in[base + 1];
        const float* Wtgt = (const float*)d_in[base + 2];
        const float* btgt = (const float*)d_in[base + 3];
        const float* avec = (const float*)d_in[base + 4];
        const float* cvec = (const float*)d_in[base + 5];
        const int K = (l == 0) ? DIN : DH;

        cvt_pack_b_kernel<<<(K * DH + 255) / 256, blk, 0, stream>>>(Wsrc, wS, K, DH);
        cvt_pack_b_kernel<<<(K * DH + 255) / 256, blk, 0, stream>>>(Wtgt, wT, K, DH);

        gemm_bf16_wmma_kernel<<<gemmGrid, blk, 0, stream>>>(hbf, wS, bsrc, xs, NN, DH, K, 0);
        gemm_bf16_wmma_kernel<<<gemmGrid, blk, 0, stream>>>(hbf, wT, btgt, xt, NN, DH, K, 0);

        seg_init_kernel<<<gNDH, blk, 0, stream>>>(mmax, ssum, agg);
        edge_score_kernel<<<gEdgeW, blk, 0, stream>>>(ei, xs, xt, avec, esc, mmax);
        edge_exp_kernel<<<gEdgeT, blk, 0, stream>>>(ei, esc, mmax, exb, ssum);
        edge_aggregate_kernel<<<gEdgeW, blk, 0, stream>>>(ei, exb, ssum, xs, agg);
        tanh_cvt_kernel<<<gNDH, blk, 0, stream>>>(agg, cvec, hbf);
    }

    // MLP head: leaky(h@Wh1+bh1), leaky(h@Wh2+bh2), then scalar projection
    cvt_pack_b_kernel<<<(DH * DH + 255) / 256, blk, 0, stream>>>(Wh1, wS, DH, DH);
    gemm_bf16_wmma_kernel<<<gemmGrid, blk, 0, stream>>>(hbf, wS, bh1, xs, NN, DH, DH, 1);
    cvt_bf16_kernel<<<gNDH, blk, 0, stream>>>(xs, hbf, nNDH);

    cvt_pack_b_kernel<<<(DH * DH + 255) / 256, blk, 0, stream>>>(Wh2, wS, DH, DH);
    gemm_bf16_wmma_kernel<<<gemmGrid, blk, 0, stream>>>(hbf, wS, bh2, xt, NN, DH, DH, 1);

    final_dot_kernel<<<(NN + 7) / 8, blk, 0, stream>>>(xt, Wh3, bh3, out);
}
